// Seq2Seq_nlpmixing_63239098466497
// MI455X (gfx1250) — matrix operand order repaired and checked by reading, not compiled, not run on hardware
//
#include <hip/hip_runtime.h>
#include <hip/hip_bf16.h>

// ---------------------------------------------------------------------------
// Seq2Seq LSTM (2-layer encoder, 2-cell decoder) fully fused per batch tile.
// B=512, S=1024, T=512, D=3, H=64, G=4H=256.
// One workgroup = one 16-row batch tile, 16 waves (512 threads), wave w owns
// gate columns [16w, 16w+16). All matrix math via v_wmma_f32_16x16x32_f16.
// A-fragment LDS loads are hoisted ahead of each WMMA chain so the ds_loads
// pipeline instead of paying full LDS latency per K-slice.
// ---------------------------------------------------------------------------

#define BATCH 512
#define SLEN  1024
#define TLEN  512
#define DIN   3
#define HID   64
#define GW    256          // 4*H
#define BT    16           // batch tile (WMMA M)
#define SA1   104          // LDS row stride (halves) for A1 = [h(64)|x(3)|0..] K=96
#define SA2   136          // LDS row stride (halves) for A2 = [h2(64)|y1(64)] K=128
#define GS    260          // LDS row stride (floats) for gate buffer

typedef __attribute__((ext_vector_type(16))) _Float16 v16h;
typedef __attribute__((ext_vector_type(8)))  _Float16 v8h;
typedef __attribute__((ext_vector_type(8)))  float    v8f;

union V16U { v16h v; v8h h[2]; };

static __device__ __forceinline__ v8f wmma_f16(v16h a, v16h b, v8f c) {
    return __builtin_amdgcn_wmma_f32_16x16x32_f16(
        /*neg_a=*/false, a, /*neg_b=*/false, b,
        /*c_mod=*/(short)0, c, /*reuse_a=*/false, /*reuse_b=*/false);
}

// Load a 16x32 f16 A fragment from LDS (row-major rows, stride in halves).
// Layout: lanes 0-15 hold row M=lane, K = kb+[0..15];
// lanes 16-31 hold row M=lane-16, K = kb+[16..31].
static __device__ __forceinline__ v16h load_A(const _Float16* base, int stride,
                                              int kb, int lane) {
    const int m  = lane & 15;
    const int ho = (lane & 16) ? 16 : 0;
    const _Float16* p = base + m * stride + kb + ho;
    v8h lo = *(const v8h*)(p);
    v8h hi = *(const v8h*)(p + 8);
    V16U a;
#pragma unroll
    for (int i = 0; i < 8; ++i) { a.h[0][i] = lo[i]; a.h[1][i] = hi[i]; }
    return a.v;
}

// ---- B fragment builders (one-time, from global f32 weights) --------------
// B layout (32x16 KxN): lane n = N (mod 16); lanes 0-15 hold K = kb+0..15,
// lanes 16-31 hold K = kb+16..31, contiguous along the 8 VGPRs.

// Combined K=96: rows 0..63 = Whh^T (Whh is [256,64] row-major),
// rows 64..66 = Wih^T (Wih is [256,3] row-major), rest zero.
static __device__ __forceinline__ v16h frag_c96(const float* __restrict__ Whh,
                                                const float* __restrict__ Wih,
                                                int nb, int kb, int lane) {
    const int n  = nb + (lane & 15);
    const int k0 = kb + ((lane & 16) ? 16 : 0);
    V16U f;
#pragma unroll
    for (int i = 0; i < 16; ++i) {
        int k = k0 + i;
        float v;
        if (k < 64)      v = Whh[n * 64 + k];
        else if (k < 67) v = Wih[n * 3 + (k - 64)];
        else             v = 0.0f;
        f.v[i] = (_Float16)v;
    }
    return f.v;
}

// Combined K=128: rows 0..63 = Whh^T, rows 64..127 = Wih^T (both [256,64]).
static __device__ __forceinline__ v16h frag_c128(const float* __restrict__ Whh,
                                                 const float* __restrict__ Wih,
                                                 int nb, int kb, int lane) {
    const int n  = nb + (lane & 15);
    const int k0 = kb + ((lane & 16) ? 16 : 0);
    V16U f;
#pragma unroll
    for (int i = 0; i < 16; ++i) {
        int k = k0 + i;
        float v = (k < 64) ? Whh[n * 64 + k] : Wih[n * 64 + (k - 64)];
        f.v[i] = (_Float16)v;
    }
    return f.v;
}

// K=64 summed weights (decoder cell 2 sees x == h): W = Wih + Whh.
static __device__ __forceinline__ v16h frag_sum(const float* __restrict__ Wih,
                                                const float* __restrict__ Whh,
                                                int nb, int kb, int lane) {
    const int n  = nb + (lane & 15);
    const int k0 = kb + ((lane & 16) ? 16 : 0);
    V16U f;
#pragma unroll
    for (int i = 0; i < 16; ++i) {
        int k = k0 + i;
        f.v[i] = (_Float16)(Wih[n * 64 + k] + Whh[n * 64 + k]);
    }
    return f.v;
}

// Scatter a C/D fragment into the gate buffer.
// C layout: VGPR r, lanes 0-15 -> row r col lane; lanes 16-31 -> row r+8.
static __device__ __forceinline__ void store_gates(float* sG, v8f c,
                                                   int nb, int lane) {
    const int n  = nb + (lane & 15);
    const int mo = (lane & 16) ? 8 : 0;
#pragma unroll
    for (int r = 0; r < 8; ++r) sG[(r + mo) * GS + n] = c[r];
}

static __device__ __forceinline__ float sigm(float x) {
    return 1.0f / (1.0f + __expf(-x));
}
static __device__ __forceinline__ float tanhx(float x) {
    float e = __expf(-2.0f * x);
    return (1.0f - e) / (1.0f + e);
}

// LSTM pointwise update for 2 cells owned by this thread (row m, cols j0,j0+1).
static __device__ __forceinline__ void lstm_update2(const float* sG,
                                                    float c[2], float h[2],
                                                    int m, int j0) {
    const float* row = sG + m * GS;
#pragma unroll
    for (int i = 0; i < 2; ++i) {
        float gi = row[j0 + i];
        float gf = row[64 + j0 + i];
        float gg = row[128 + j0 + i];
        float go = row[192 + j0 + i];
        float cn = sigm(gf) * c[i] + sigm(gi) * tanhx(gg);
        c[i] = cn;
        h[i] = sigm(go) * tanhx(cn);
    }
}

__global__ __launch_bounds__(512, 1)
void seq2seq_lstm_kernel(const float* __restrict__ src, const float* __restrict__ trg,
                         const float* __restrict__ e1Wih, const float* __restrict__ e1Whh,
                         const float* __restrict__ e1b,
                         const float* __restrict__ e2Wih, const float* __restrict__ e2Whh,
                         const float* __restrict__ e2b,
                         const float* __restrict__ d1Wih, const float* __restrict__ d1Whh,
                         const float* __restrict__ d1b,
                         const float* __restrict__ d2Wih, const float* __restrict__ d2Whh,
                         const float* __restrict__ d2b,
                         const float* __restrict__ fcW, const float* __restrict__ fcb,
                         float* __restrict__ out) {
    __shared__ __align__(16) _Float16 sA1[BT * SA1];  // [h | x | 0] for D-input cells
    __shared__ __align__(16) _Float16 sA2[BT * SA2];  // [h2 | y1] / [h1] staging
    __shared__ __align__(16) float    sG [BT * GS];   // gate pre-activations
    __shared__ __align__(16) float    sH2[BT * HID];  // decoder h2 (f32) for fc

    const int t    = threadIdx.x;
    const int lane = t & 31;
    const int nb   = (t >> 5) * 16;       // this wave's gate-column base
    const int b0   = blockIdx.x * BT;     // batch-row base

    // Cell ownership: 2 (row, col) cells per thread, contiguous cols.
    const int cm = t >> 5;                // == (2t)/64, row 0..15
    const int cj = (t * 2) & 63;          // col j0

    // ---- init LDS (h = 0, padding = 0) ----
    for (int i = t; i < BT * SA1; i += blockDim.x) sA1[i] = (_Float16)0.0f;
    for (int i = t; i < BT * SA2; i += blockDim.x) sA2[i] = (_Float16)0.0f;
    // stage x_0 = src[:, 0]
    if (t < BT * DIN) {
        int m = t / DIN, cc = t % DIN;
        sA1[m * SA1 + 64 + cc] =
            (_Float16)src[((size_t)(b0 + m) * SLEN + 0) * DIN + cc];
    }

    // ---- build encoder weight fragments (register-resident) ----
    v16h B1[3], B2[4];
#pragma unroll
    for (int ks = 0; ks < 3; ++ks) B1[ks] = frag_c96 (e1Whh, e1Wih, nb, ks * 32, lane);
#pragma unroll
    for (int ks = 0; ks < 4; ++ks) B2[ks] = frag_c128(e2Whh, e2Wih, nb, ks * 32, lane);
    const float b1v = e1b[nb + (lane & 15)];
    const float b2v = e2b[nb + (lane & 15)];

    float c1[2] = {0.0f, 0.0f};
    float c2[2] = {0.0f, 0.0f};
    float hbuf[2] = {0.0f, 0.0f};

    __syncthreads();

    // ======================= ENCODER: S steps ==============================
    for (int step = 0; step < SLEN; ++step) {
        // layer 1: gates1 = [h1|x] @ B1 + b1   (K = 96)
        // Hoist all A-slice loads so the ds_loads pipeline.
        v16h a1f[3];
#pragma unroll
        for (int ks = 0; ks < 3; ++ks) a1f[ks] = load_A(sA1, SA1, ks * 32, lane);
        v8f acc = {b1v, b1v, b1v, b1v, b1v, b1v, b1v, b1v};
#pragma unroll
        for (int ks = 0; ks < 3; ++ks) acc = wmma_f16(a1f[ks], B1[ks], acc);
        store_gates(sG, acc, nb, lane);
        __syncthreads();

        // update1: h1 -> sA1[0..63] (own recurrence) and sA2[64..127] (y1)
        lstm_update2(sG, c1, hbuf, cm, cj);
        sA1[cm * SA1 + cj]          = (_Float16)hbuf[0];
        sA1[cm * SA1 + cj + 1]      = (_Float16)hbuf[1];
        sA2[cm * SA2 + 64 + cj]     = (_Float16)hbuf[0];
        sA2[cm * SA2 + 64 + cj + 1] = (_Float16)hbuf[1];
        // stage x_{t+1} (touches only cols 64..66 of sA1)
        if (step + 1 < SLEN && t < BT * DIN) {
            int m = t / DIN, cc = t % DIN;
            sA1[m * SA1 + 64 + cc] =
                (_Float16)src[((size_t)(b0 + m) * SLEN + step + 1) * DIN + cc];
        }
        __syncthreads();

        // layer 2: gates2 = [h2|y1] @ B2 + b2   (K = 128)
        v16h a2f[4];
#pragma unroll
        for (int ks = 0; ks < 4; ++ks) a2f[ks] = load_A(sA2, SA2, ks * 32, lane);
        v8f acc2 = {b2v, b2v, b2v, b2v, b2v, b2v, b2v, b2v};
#pragma unroll
        for (int ks = 0; ks < 4; ++ks) acc2 = wmma_f16(a2f[ks], B2[ks], acc2);
        store_gates(sG, acc2, nb, lane);
        __syncthreads();

        // update2: h2 -> sA2[0..63]
        lstm_update2(sG, c2, hbuf, cm, cj);
        sA2[cm * SA2 + cj]     = (_Float16)hbuf[0];
        sA2[cm * SA2 + cj + 1] = (_Float16)hbuf[1];
        __syncthreads();
    }

    // ================= encoder -> decoder handoff ==========================
    // decoder state: h = enc2 final h (in hbuf), c = enc2 final c (in c2 regs)
    sA1[cm * SA1 + cj]     = (_Float16)hbuf[0];
    sA1[cm * SA1 + cj + 1] = (_Float16)hbuf[1];
    if (t < BT * DIN) {
        int m = t / DIN, cc = t % DIN;
        sA1[m * SA1 + 64 + cc] =
            (_Float16)trg[((size_t)(b0 + m) * TLEN + 0) * DIN + cc];
        out[((size_t)(b0 + m) * TLEN + 0) * DIN + cc] = 0.0f;  // outputs[:,0] = 0
    }

    // decoder weight fragments
    v16h Bd1[3], Bs[2];
#pragma unroll
    for (int ks = 0; ks < 3; ++ks) Bd1[ks] = frag_c96(d1Whh, d1Wih, nb, ks * 32, lane);
#pragma unroll
    for (int ks = 0; ks < 2; ++ks) Bs[ks]  = frag_sum(d2Wih, d2Whh, nb, ks * 32, lane);
    const float bd1 = d1b[nb + (lane & 15)];
    const float bd2 = d2b[nb + (lane & 15)];

    float cd[2] = {c2[0], c2[1]};
    __syncthreads();

    // ======================= DECODER: T-1 steps ============================
    for (int s = 0; s < TLEN - 1; ++s) {
        // cell 1: gates = [h|x_s] @ Bd1 + bd1   (K = 96)
        v16h ad1[3];
#pragma unroll
        for (int ks = 0; ks < 3; ++ks) ad1[ks] = load_A(sA1, SA1, ks * 32, lane);
        v8f a1 = {bd1, bd1, bd1, bd1, bd1, bd1, bd1, bd1};
#pragma unroll
        for (int ks = 0; ks < 3; ++ks) a1 = wmma_f16(ad1[ks], Bd1[ks], a1);
        store_gates(sG, a1, nb, lane);
        __syncthreads();

        // update cell 1: (h,c) -> (h1,c1); h1 -> sA2[0..63]
        float cl[2] = {cd[0], cd[1]};
        lstm_update2(sG, cl, hbuf, cm, cj);
        sA2[cm * SA2 + cj]     = (_Float16)hbuf[0];
        sA2[cm * SA2 + cj + 1] = (_Float16)hbuf[1];
        if (s + 1 < TLEN - 1 && t < BT * DIN) {
            int m = t / DIN, cc = t % DIN;
            sA1[m * SA1 + 64 + cc] =
                (_Float16)trg[((size_t)(b0 + m) * TLEN + s + 1) * DIN + cc];
        }
        __syncthreads();

        // cell 2: x = h = h1 -> single K=64 GEMM with summed weights
        v16h ad2[2];
#pragma unroll
        for (int ks = 0; ks < 2; ++ks) ad2[ks] = load_A(sA2, SA2, ks * 32, lane);
        v8f a2 = {bd2, bd2, bd2, bd2, bd2, bd2, bd2, bd2};
#pragma unroll
        for (int ks = 0; ks < 2; ++ks) a2 = wmma_f16(ad2[ks], Bs[ks], a2);
        store_gates(sG, a2, nb, lane);
        __syncthreads();

        // update cell 2: (h1,c1) -> (h2,c2); carry; h2 -> sA1 and sH2
        lstm_update2(sG, cl, hbuf, cm, cj);
        cd[0] = cl[0]; cd[1] = cl[1];
        sA1[cm * SA1 + cj]     = (_Float16)hbuf[0];
        sA1[cm * SA1 + cj + 1] = (_Float16)hbuf[1];
        sH2[cm * HID + cj]     = hbuf[0];
        sH2[cm * HID + cj + 1] = hbuf[1];
        __syncthreads();

        // fc: out[:, s+1] = h2 @ fcW^T + fcb   (48 threads; others start WMMA)
        if (t < BT * DIN) {
            int m = t / DIN, cc = t % DIN;
            float a = fcb[cc];
#pragma unroll 8
            for (int j = 0; j < HID; ++j) a += sH2[m * HID + j] * fcW[cc * HID + j];
            out[((size_t)(b0 + m) * TLEN + (s + 1)) * DIN + cc] = a;
        }
    }
}

extern "C" void kernel_launch(void* const* d_in, const int* in_sizes, int n_in,
                              void* d_out, int out_size, void* d_ws, size_t ws_size,
                              hipStream_t stream) {
    (void)in_sizes; (void)n_in; (void)out_size; (void)d_ws; (void)ws_size;
    const float* src   = (const float*)d_in[0];
    const float* trg   = (const float*)d_in[1];
    const float* e1Wih = (const float*)d_in[2];
    const float* e1Whh = (const float*)d_in[3];
    const float* e1b   = (const float*)d_in[4];
    const float* e2Wih = (const float*)d_in[5];
    const float* e2Whh = (const float*)d_in[6];
    const float* e2b   = (const float*)d_in[7];
    const float* d1Wih = (const float*)d_in[8];
    const float* d1Whh = (const float*)d_in[9];
    const float* d1b   = (const float*)d_in[10];
    const float* d2Wih = (const float*)d_in[11];
    const float* d2Whh = (const float*)d_in[12];
    const float* d2b   = (const float*)d_in[13];
    const float* fcW   = (const float*)d_in[14];
    const float* fcb   = (const float*)d_in[15];
    float* out = (float*)d_out;

    seq2seq_lstm_kernel<<<dim3(BATCH / BT), dim3(512), 0, stream>>>(
        src, trg, e1Wih, e1Whh, e1b, e2Wih, e2Whh, e2b,
        d1Wih, d1Whh, d1b, d2Wih, d2Whh, d2b, fcW, fcb, out);
}
